// AdaWarp_29016799052668
// MI455X (gfx1250) — matrix operand order repaired and checked by reading, not compiled, not run on hardware
//
#include <hip/hip_runtime.h>

#define Bn  4
#define Cn  32
#define Hn  256
#define Wn  256
#define NFn 9
#define HWn (Hn * Wn)
#define PS  36   // padded per-pixel channel stride (elements); 72B stride -> conflict-free banks

typedef __attribute__((ext_vector_type(16))) __bf16 v16bf;
typedef __attribute__((ext_vector_type(4)))  __bf16 v4bf;
typedef __attribute__((ext_vector_type(8)))  float  v8f;
typedef __attribute__((ext_vector_type(4)))  float  v4f;

// pack 32 f32 -> 32 bf16 (v_cvt_pk_bf16_f32) and store as ds_store_b64s
__device__ inline void store_bf16x32(__bf16* __restrict__ dst, const float* __restrict__ v) {
#pragma unroll
  for (int j = 0; j < 8; ++j) {
    v4bf t;
    t[0] = (__bf16)v[4 * j + 0];
    t[1] = (__bf16)v[4 * j + 1];
    t[2] = (__bf16)v[4 * j + 2];
    t[3] = (__bf16)v[4 * j + 3];
    *(v4bf*)(dst + 4 * j) = t;
  }
}

// A-fragment: rows [rowbase..rowbase+15] of a row-major 32x32 f32 weight matrix.
// ISA 7.12.2 16-bit A 16x32: lanes 0-15 -> K=0..7 (elems 0..7), K=16..23 (elems 8..15);
// lanes 16-31 -> K=8..15 / 24..31.
__device__ inline v16bf load_w_frag(const float* __restrict__ Wmat, int rowbase, int lane) {
  int m  = rowbase + (lane & 15);
  int kb = (lane < 16) ? 0 : 8;
  v16bf a;
#pragma unroll
  for (int i = 0; i < 16; ++i) {
    int k = kb + i + ((i >= 8) ? 8 : 0);
    a[i] = (__bf16)Wmat[m * 32 + k];
  }
  return a;
}

// B-fragment: 32(K) x 16(N) from pixel-major LDS ([pixel][PS] bf16).
// Lanes 0-15 hold K=0..7,16..23; lanes 16-31 hold K=8..15,24..31; N = lane&15.
__device__ inline v16bf load_b_frag_pm(const __bf16* __restrict__ S, int p, int lane) {
  const __bf16* base = S + p * PS + ((lane < 16) ? 0 : 8);
  v16bf b;
#pragma unroll
  for (int j = 0; j < 4; ++j) {
    v4bf t = *(const v4bf*)(base + 4 * j + ((j >= 2) ? 8 : 0));
    b[4 * j + 0] = t[0];
    b[4 * j + 1] = t[1];
    b[4 * j + 2] = t[2];
    b[4 * j + 3] = t[3];
  }
  return b;
}

// D-fragment + bias -> pixel-major f32 LDS, 2x ds_store_b128 per lane
__device__ inline void store_d_pm(float* __restrict__ dst, int p, int ebase, v8f acc,
                                  const float* __restrict__ bias) {
  v4f lo, hi;
#pragma unroll
  for (int r = 0; r < 4; ++r) lo[r] = acc[r] + bias[ebase + r];
#pragma unroll
  for (int r = 0; r < 4; ++r) hi[r] = acc[4 + r] + bias[ebase + 4 + r];
  *(v4f*)(dst + p * PS + ebase)     = lo;
  *(v4f*)(dst + p * PS + ebase + 4) = hi;
}

// ---------------- Kernel 1: warp-sample + attention + out_proj + residual ----
__global__ __launch_bounds__(128)
void attn_kernel(const float* __restrict__ x, const float* __restrict__ flows,
                 const float* __restrict__ masks, const float* __restrict__ pos_enc,
                 const float* __restrict__ ipw, const float* __restrict__ ipb,
                 const float* __restrict__ opw, const float* __restrict__ opb,
                 float* __restrict__ outbuf) {
  __shared__ __bf16 xs[128 * PS];            //  9.2 KB (x tile, later ctx)
  __shared__ __bf16 smp[NFn][128 * PS];      // 82.9 KB (all sampled frames)
  __shared__ float  qs[128 * PS];            // 18.4 KB
  __shared__ float  ks[2][128 * PS];         // 36.9 KB k/v ping-pong (1 barrier/frame)
  __shared__ float  sc[NFn * 128];           //  4.6 KB scores -> attn weights

  const int tid  = threadIdx.x;
  const int lane = tid & 31;
  const int wid  = tid >> 5;
  const int bid  = blockIdx.x;
  const int b    = bid / (Hn * 2);
  const int rem  = bid % (Hn * 2);
  const int h    = rem >> 1;
  const int w0   = (rem & 1) * 128;
  const int wg   = w0 + tid;

  const float* xb = x + (size_t)b * Cn * HWn;

  // stage x row-tile (pixel-major bf16)
  {
    float v[Cn];
#pragma unroll
    for (int c = 0; c < Cn; ++c) v[c] = xb[c * HWn + h * Wn + wg];
    store_bf16x32(xs + tid * PS, v);
  }

  // bilinear-sample every frame once into LDS (+pos_enc)
  for (int n = 0; n < NFn; ++n) {
    float mv = masks[((size_t)(b * NFn + n)) * HWn + h * Wn + wg];
    float fx = flows[((size_t)((b * 2 + 0) * NFn + n)) * HWn + h * Wn + wg] * mv;
    float fy = flows[((size_t)((b * 2 + 1) * NFn + n)) * HWn + h * Wn + wg] * mv;
    float px = (float)wg + fx, py = (float)h + fy;
    float x0f = floorf(px), y0f = floorf(py);
    float wx1 = px - x0f, wx0 = 1.f - wx1;
    float wy1 = py - y0f, wy0 = 1.f - wy1;
    int xi0 = (int)x0f, yi0 = (int)y0f, xi1 = xi0 + 1, yi1 = yi0 + 1;
    float v0x = (xi0 >= 0 && xi0 < Wn) ? 1.f : 0.f;
    float v1x = (xi1 >= 0 && xi1 < Wn) ? 1.f : 0.f;
    float v0y = (yi0 >= 0 && yi0 < Hn) ? 1.f : 0.f;
    float v1y = (yi1 >= 0 && yi1 < Hn) ? 1.f : 0.f;
    int xc0 = min(max(xi0, 0), Wn - 1), xc1 = min(max(xi1, 0), Wn - 1);
    int yc0 = min(max(yi0, 0), Hn - 1), yc1 = min(max(yi1, 0), Hn - 1);
    float c00 = wx0 * wy0 * v0x * v0y, c10 = wx1 * wy0 * v1x * v0y;
    float c01 = wx0 * wy1 * v0x * v1y, c11 = wx1 * wy1 * v1x * v1y;
    int i00 = yc0 * Wn + xc0, i10 = yc0 * Wn + xc1;
    int i01 = yc1 * Wn + xc0, i11 = yc1 * Wn + xc1;
    float pe = pos_enc[n];
    float s[Cn];
#pragma unroll
    for (int c = 0; c < Cn; ++c) {
      const float* pc = xb + c * HWn;
      s[c] = c00 * pc[i00] + c10 * pc[i10] + c01 * pc[i01] + c11 * pc[i11] + pe;
    }
    store_bf16x32(&smp[n][tid * PS], s);
  }
  __syncthreads();

  // fixed M-tile per wave -> single hoisted A-fragment per projection
  const int mt    = wid & 1;
  const int ntb   = (wid >> 1) * 4;
  const int ebase = mt * 16 + ((lane < 16) ? 0 : 8);
  const int pln   = lane & 15;

  // Q = Wq * X + bq
  {
    v16bf aQ = load_w_frag(ipw, mt * 16, lane);
#pragma unroll
    for (int i = 0; i < 4; ++i) {
      int p = (ntb + i) * 16 + pln;
      v16bf bb = load_b_frag_pm(xs, p, lane);
      v8f acc = {};
      acc = __builtin_amdgcn_wmma_f32_16x16x32_bf16(false, aQ, false, bb, (short)0, acc, false, false);
      store_d_pm(qs, p, ebase, acc, ipb);
    }
  }
  __syncthreads();

  const float scale = 0.17677669529663687f;   // 1/sqrt(32)
  {
    v16bf aK = load_w_frag(ipw + Cn * Cn, mt * 16, lane);
    for (int n = 0; n < NFn; ++n) {            // rolled: caps VGPR pressure
      float* kbuf = ks[n & 1];
#pragma unroll
      for (int i = 0; i < 4; ++i) {
        int p = (ntb + i) * 16 + pln;
        v16bf bb = load_b_frag_pm(smp[n], p, lane);
        v8f acc = {};
        acc = __builtin_amdgcn_wmma_f32_16x16x32_bf16(false, aK, false, bb, (short)0, acc, false, false);
        store_d_pm(kbuf, p, ebase, acc, ipb + Cn);
      }
      __syncthreads();                         // one barrier per frame (ping-pong)
      float s = 0.f;
      const v4f* qp = (const v4f*)(qs + tid * PS);
      const v4f* kp = (const v4f*)(kbuf + tid * PS);
#pragma unroll
      for (int j = 0; j < 8; ++j) {
        v4f a4 = qp[j], b4 = kp[j];
        s += a4[0] * b4[0] + a4[1] * b4[1] + a4[2] * b4[2] + a4[3] * b4[3];
      }
      sc[n * 128 + tid] = s * scale;
    }
  }

  // softmax over NF (per-thread column of sc; no cross-thread sharing)
  {
    float scv[NFn];
#pragma unroll
    for (int n = 0; n < NFn; ++n) scv[n] = sc[n * 128 + tid];
    float mx = scv[0];
#pragma unroll
    for (int n = 1; n < NFn; ++n) mx = fmaxf(mx, scv[n]);
    float ssum = 0.f;
#pragma unroll
    for (int n = 0; n < NFn; ++n) { scv[n] = __expf(scv[n] - mx); ssum += scv[n]; }
    float inv = 1.f / ssum;
#pragma unroll
    for (int n = 0; n < NFn; ++n) sc[n * 128 + tid] = scv[n] * inv;
  }

  float ctx[Cn];
#pragma unroll
  for (int e = 0; e < Cn; ++e) ctx[e] = 0.f;

  {
    v16bf aV = load_w_frag(ipw + 2 * Cn * Cn, mt * 16, lane);
    for (int n = 0; n < NFn; ++n) {            // rolled + ping-pong
      float* vbuf = ks[n & 1];
#pragma unroll
      for (int i = 0; i < 4; ++i) {
        int p = (ntb + i) * 16 + pln;
        v16bf bb = load_b_frag_pm(smp[n], p, lane);
        v8f acc = {};
        acc = __builtin_amdgcn_wmma_f32_16x16x32_bf16(false, aV, false, bb, (short)0, acc, false, false);
        store_d_pm(vbuf, p, ebase, acc, ipb + 2 * Cn);
      }
      __syncthreads();
      float an = sc[n * 128 + tid];
      const v4f* vp = (const v4f*)(vbuf + tid * PS);
#pragma unroll
      for (int j = 0; j < 8; ++j) {
        v4f v4 = vp[j];
        ctx[4 * j + 0] += an * v4[0];
        ctx[4 * j + 1] += an * v4[1];
        ctx[4 * j + 2] += an * v4[2];
        ctx[4 * j + 3] += an * v4[3];
      }
    }
  }
  __syncthreads();                             // xs readers (Q) long done; reuse for ctx

  store_bf16x32(xs + tid * PS, ctx);
  __syncthreads();

  // out = Wo * ctx + bo + x  (f32 to workspace)
  {
    v16bf aO = load_w_frag(opw, mt * 16, lane);
#pragma unroll
    for (int i = 0; i < 4; ++i) {
      int p = (ntb + i) * 16 + pln;
      v16bf bb = load_b_frag_pm(xs, p, lane);
      v8f acc = {};
      acc = __builtin_amdgcn_wmma_f32_16x16x32_bf16(false, aO, false, bb, (short)0, acc, false, false);
#pragma unroll
      for (int r = 0; r < 8; ++r) {
        int e = ebase + r;
        size_t idx = ((size_t)(b * Cn + e)) * HWn + (size_t)h * Wn + (w0 + p);
        outbuf[idx] = acc[r] + opb[e] + x[idx];
      }
    }
  }
}

// ---------------- Kernel 2: 3x3 conv (implicit GEMM) + LeakyReLU + residual --
__global__ __launch_bounds__(256)
void conv_kernel(const float* __restrict__ inb, const float* __restrict__ cw,
                 const float* __restrict__ cb, float* __restrict__ outp) {
  __shared__ __bf16 ins[3 * 258 * PS];       // 55.7 KB, pixel-major, zero-padded cols
  const int tid  = threadIdx.x;
  const int lane = tid & 31;
  const int wid  = tid >> 5;
  const int b    = blockIdx.x / Hn;
  const int h    = blockIdx.x % Hn;

  for (int r = 0; r < 3; ++r) {
    int hh = h + r - 1;
    bool ok = (hh >= 0) && (hh < Hn);
    float v[Cn];
#pragma unroll
    for (int c = 0; c < Cn; ++c)
      v[c] = ok ? inb[((size_t)(b * Cn + c)) * HWn + (size_t)hh * Wn + tid] : 0.f;
    store_bf16x32(ins + (r * 258 + tid + 1) * PS, v);
  }
  if (tid < 6) {
    float z[Cn];
#pragma unroll
    for (int c = 0; c < Cn; ++c) z[c] = 0.f;
    int r = tid >> 1, col = (tid & 1) * 257;
    store_bf16x32(ins + (r * 258 + col) * PS, z);
  }
  __syncthreads();

  const int mt    = wid & 1;
  const int ntb   = (wid >> 1) * 4;
  const int ebase = mt * 16 + ((lane < 16) ? 0 : 8);
  const int pln   = lane & 15;

  // hoist all 9 tap A-fragments (fixed mt per wave)
  v16bf aw[9];
#pragma unroll
  for (int kh = 0; kh < 3; ++kh) {
#pragma unroll
    for (int kw = 0; kw < 3; ++kw) {
      int m  = mt * 16 + (lane & 15);
      int kb = (lane < 16) ? 0 : 8;
      v16bf a;
#pragma unroll
      for (int e = 0; e < 16; ++e) {
        int k = kb + e + ((e >= 8) ? 8 : 0);
        a[e] = (__bf16)cw[((m * Cn + k) * 3 + kh) * 3 + kw];
      }
      aw[kh * 3 + kw] = a;
    }
  }

  for (int i = 0; i < 4; ++i) {                // rolled tile loop (pressure cap)
    int p = (ntb + i) * 16 + pln;
    v8f acc = {};
#pragma unroll
    for (int kh = 0; kh < 3; ++kh) {
#pragma unroll
      for (int kw = 0; kw < 3; ++kw) {
        v16bf bb = load_b_frag_pm(ins + (kh * 258) * PS, p + kw, lane);
        acc = __builtin_amdgcn_wmma_f32_16x16x32_bf16(false, aw[kh * 3 + kw], false, bb,
                                                      (short)0, acc, false, false);
      }
    }
#pragma unroll
    for (int r = 0; r < 8; ++r) {
      int o = ebase + r;
      size_t idx = ((size_t)(b * Cn + o)) * HWn + (size_t)h * Wn + p;
      float cv  = acc[r] + cb[o];
      float ffn = (cv >= 0.f) ? cv : 0.1f * cv;
      outp[idx] = ffn + inb[idx];
    }
  }
}

extern "C" void kernel_launch(void* const* d_in, const int* in_sizes, int n_in,
                              void* d_out, int out_size, void* d_ws, size_t ws_size,
                              hipStream_t stream) {
  const float* x          = (const float*)d_in[0];
  const float* flows      = (const float*)d_in[1];
  const float* masks      = (const float*)d_in[2];
  const float* pos_enc    = (const float*)d_in[3];
  const float* in_proj_w  = (const float*)d_in[4];
  const float* in_proj_b  = (const float*)d_in[5];
  const float* out_proj_w = (const float*)d_in[6];
  const float* out_proj_b = (const float*)d_in[7];
  const float* conv_w     = (const float*)d_in[8];
  const float* conv_b     = (const float*)d_in[9];

  float* outbuf    = (float*)d_ws;   // B*C*H*W f32 intermediate (33.5 MB)
  float* final_out = (float*)d_out;

  attn_kernel<<<dim3(Bn * Hn * 2), dim3(128), 0, stream>>>(
      x, flows, masks, pos_enc, in_proj_w, in_proj_b, out_proj_w, out_proj_b, outbuf);
  conv_kernel<<<dim3(Bn * Hn), dim3(256), 0, stream>>>(
      outbuf, conv_w, conv_b, final_out);
}
